// LSTMEncoder_42408507081014
// MI455X (gfx1250) — matrix operand order repaired and checked by reading, not compiled, not run on hardware
//
#include <hip/hip_runtime.h>
#include <hip/hip_bf16.h>

// ---------------------------------------------------------------------------
// CDNA5 (gfx1250) implementation.
//
// All matmuls are lowered to one batched WMMA kernel computing
//   C[b] = A[b](M,K) @ Bt[b](N,K)^T      (bf16 in, f32 accumulate)
// because v_wmma_f32_16x16x32_bf16 wants both operands "K-contiguous within a
// lane" -- i.e. row-major-by-K for A and for B-transposed.
//
// Workspace layout (bytes, 256-aligned; ~238 MB total):
//   sn    : normalized src, bf16 (32768 x 320)          20,971,520
//   tn    : normalized table, bf16 (32*256 x 320)        5,242,880
//   schn  : normalized schema, bf16 (32 x 128pad x 320)  2,621,440
//   wf/wb : W_ih bf16 (640pad x 320)                     2 x 409,600
//   whf/whb: W_hh bf16 (640pad x 160)                    2 x 204,800
//   bias  : b_ih+b_hh f32 (2 x 600)
//   norms : row L2 norms of src (32768 f32)
//   pre_f/pre_b : x@W_ih^T+bias, f32 (32768 x 600)       2 x 78,643,200
//   sim_t : masked cosine sim bf16 (32 x 256 x 1024)    16,777,216
//   sim_s : masked cosine sim bf16 (32 x 128pad x 1024)  8,388,608
//   encT  : transposed encodings bf16 (32 x 384pad x 1024) 25,165,824
// ---------------------------------------------------------------------------

typedef __attribute__((ext_vector_type(16))) __bf16 v16bf;
typedef __attribute__((ext_vector_type(8)))  float  v8f;

union Frag { v16bf v; uint4 q[2]; };

__device__ __forceinline__ unsigned short f2bf(float f) {
  unsigned int u = __float_as_uint(f);
  u = (u + 0x7FFFu + ((u >> 16) & 1u)) >> 16;   // round-to-nearest-even
  return (unsigned short)u;
}
__device__ __forceinline__ float bf2f(unsigned short h) {
  return __uint_as_float(((unsigned int)h) << 16);
}
__device__ __forceinline__ float sigmoidf_(float x) { return 1.0f / (1.0f + __expf(-x)); }

// ---------------------------------------------------------------------------
// Stage rows: f32 -> bf16, optional L2 normalization, zero padding of rows/cols.
// grid.x = batches * padRows, block = 256.
// ---------------------------------------------------------------------------
__global__ __launch_bounds__(256)
void stage_rows_kernel(const float* __restrict__ in, unsigned short* __restrict__ out,
                       float* __restrict__ norms,
                       int rowsPerB, int padRows, int Kin, int Kpad, int normalize) {
  const int r = blockIdx.x % padRows;
  const int b = blockIdx.x / padRows;
  const size_t orow = (size_t)blockIdx.x * Kpad;
  const int tid = threadIdx.x;
  if (r >= rowsPerB) {              // pure padding row
    for (int c = tid; c < Kpad; c += 256) out[orow + c] = 0;
    return;
  }
  const float* irow = in + ((size_t)b * rowsPerB + r) * Kin;
  float scale = 1.0f;
  if (normalize) {
    __shared__ float red[256];
    float s = 0.f;
    for (int c = tid; c < Kin; c += 256) { float v = irow[c]; s += v * v; }
    red[tid] = s; __syncthreads();
    for (int w = 128; w > 0; w >>= 1) { if (tid < w) red[tid] += red[tid + w]; __syncthreads(); }
    float nrm = fmaxf(sqrtf(red[0]), 1e-8f);
    scale = 1.0f / nrm;
    if (norms && tid == 0) norms[blockIdx.x] = nrm;
  }
  for (int c = tid; c < Kpad; c += 256)
    out[orow + c] = (c < Kin) ? f2bf(irow[c] * scale) : (unsigned short)0;
}

__global__ void bias_kernel(const float* bif, const float* bhf,
                            const float* bib, const float* bhb, float* out) {
  int i = blockIdx.x * blockDim.x + threadIdx.x;
  if (i < 600)       out[i] = bif[i] + bhf[i];
  else if (i < 1200) out[i] = bib[i - 600] + bhb[i - 600];
}

__global__ void zero16_kernel(uint4* p, long long n16) {
  long long i = (long long)blockIdx.x * blockDim.x + threadIdx.x;
  if (i < n16) p[i] = make_uint4(0u, 0u, 0u, 0u);
}

// ---------------------------------------------------------------------------
// Batched GEMM  C = A(M,K) @ Bt(N,K)^T  via v_wmma_f32_16x16x32_bf16.
// Block: 256 thr (8 waves, 2x4), macro-tile 128x128, K stepped by 32.
// Inputs are pre-padded in ws, so loads are unguarded 16B vectors.
// Epilogues:
//   0: f32 out = acc*rowScale[m] + bias[n]           (pre-projection)
//   1: bf16 out = mask[b,m]>0 ? 0 : acc              (cosine sim)
//   2: f32 out = acc + table_in + col_hot@col_W^T+b  (table context)
//   3: f32 out = acc + schema_in                     (schema context)
// ---------------------------------------------------------------------------
#define EPI_BIAS_F32   0
#define EPI_MASK_BF16  1
#define EPI_ADD_TABLE  2
#define EPI_ADD_SCHEMA 3

__global__ __launch_bounds__(256)
void gemm_bt_kernel(const unsigned short* __restrict__ A,
                    const unsigned short* __restrict__ Bt,
                    int K, long long sA, long long sB,
                    int epi, float* __restrict__ outF, unsigned short* __restrict__ outBF,
                    long long sC, int ldc, int Mvalid, int Nvalid,
                    const float* __restrict__ bias, const float* __restrict__ rowScale,
                    const int* __restrict__ mask, int maskStride,
                    const float* __restrict__ addIn, long long sAdd,
                    const float* __restrict__ hot, const float* __restrict__ colW,
                    const float* __restrict__ colb) {
  __shared__ __align__(16) unsigned short smA[128 * 32];
  __shared__ __align__(16) unsigned short smB[128 * 32];

  const int tid   = threadIdx.x;
  const int lane  = tid & 31;
  const int wave  = tid >> 5;
  const int waveM = wave >> 2;   // 0..1  (64 rows each)
  const int waveN = wave & 3;    // 0..3  (32 cols each)
  const int l16   = lane >> 4;
  const int ln    = lane & 15;
  const int batch = blockIdx.z;

  const unsigned short* Ab = A  + (long long)batch * sA + (size_t)blockIdx.y * 128 * K;
  const unsigned short* Bb = Bt + (long long)batch * sB + (size_t)blockIdx.x * 128 * K;

  const int lrow = tid >> 1;          // 0..127
  const int lcol = (tid & 1) * 16;    // 0 or 16

  const v8f vzero = {0.f, 0.f, 0.f, 0.f, 0.f, 0.f, 0.f, 0.f};
  v8f acc[4][2];
#pragma unroll
  for (int mt = 0; mt < 4; ++mt)
#pragma unroll
    for (int nt = 0; nt < 2; ++nt) acc[mt][nt] = vzero;

  const int KT = K >> 5;
  for (int kt = 0; kt < KT; ++kt) {
    const int k0 = kt << 5;
    const uint4* ga = (const uint4*)(Ab + (size_t)lrow * K + k0 + lcol);
    const uint4* gb = (const uint4*)(Bb + (size_t)lrow * K + k0 + lcol);
    uint4 a0 = ga[0], a1 = ga[1], b0 = gb[0], b1 = gb[1];
    if (kt + 1 < KT) {   // global_prefetch_b8 of next K slice
      __builtin_prefetch(Ab + (size_t)lrow * K + k0 + 32 + lcol, 0, 1);
      __builtin_prefetch(Bb + (size_t)lrow * K + k0 + 32 + lcol, 0, 1);
    }
    __syncthreads();
    *(uint4*)&smA[lrow * 32 + lcol]     = a0;
    *(uint4*)&smA[lrow * 32 + lcol + 8] = a1;
    *(uint4*)&smB[lrow * 32 + lcol]     = b0;
    *(uint4*)&smB[lrow * 32 + lcol + 8] = b1;
    __syncthreads();

    Frag afr[4], bfr[2];
#pragma unroll
    for (int mt = 0; mt < 4; ++mt) {           // A frag: lane m+16h holds K {8h..8h+7, 16+8h..}
      int base = (waveM * 64 + mt * 16 + ln) * 32 + 8 * l16;
      afr[mt].q[0] = *(const uint4*)&smA[base];
      afr[mt].q[1] = *(const uint4*)&smA[base + 16];
    }
#pragma unroll
    for (int nt = 0; nt < 2; ++nt) {           // B frag: lane n+16h holds K {8h..8h+7, 16+8h..}
      int base = (waveN * 32 + nt * 16 + ln) * 32 + 8 * l16;
      bfr[nt].q[0] = *(const uint4*)&smB[base];
      bfr[nt].q[1] = *(const uint4*)&smB[base + 16];
    }
#pragma unroll
    for (int mt = 0; mt < 4; ++mt)
#pragma unroll
      for (int nt = 0; nt < 2; ++nt)
        acc[mt][nt] = __builtin_amdgcn_wmma_f32_16x16x32_bf16(
            false, afr[mt].v, false, bfr[nt].v, (short)0, acc[mt][nt], false, false);
  }

  // Epilogue.  C layout: VGPR r -> m = r + 8*(lane>=16), n = lane&15.
#pragma unroll
  for (int mt = 0; mt < 4; ++mt) {
#pragma unroll
    for (int nt = 0; nt < 2; ++nt) {
#pragma unroll
      for (int r = 0; r < 8; ++r) {
        const int gm = blockIdx.y * 128 + waveM * 64 + mt * 16 + r + 8 * l16;
        const int gn = blockIdx.x * 128 + waveN * 32 + nt * 16 + ln;
        float v = acc[mt][nt][r];
        if (epi == EPI_BIAS_F32) {
          if (gm < Mvalid && gn < Nvalid)
            outF[(long long)batch * sC + (size_t)gm * ldc + gn] = v * rowScale[gm] + bias[gn];
        } else if (epi == EPI_MASK_BF16) {
          int mv = (gm < Mvalid) ? mask[(size_t)batch * maskStride + gm] : 1;
          if (gn < Nvalid)
            outBF[(long long)batch * sC + (size_t)gm * ldc + gn] = f2bf(mv > 0 ? 0.0f : v);
        } else if (epi == EPI_ADD_TABLE) {
          if (gm < Mvalid && gn < Nvalid) {
            const float* h9 = hot + ((size_t)batch * Mvalid + gm) * 9;
            const float* w9 = colW + (size_t)gn * 9;
            float ct = colb[gn];
#pragma unroll
            for (int j = 0; j < 9; ++j) ct += h9[j] * w9[j];
            long long o = (long long)batch * sC + (size_t)gm * ldc + gn;
            outF[o] = v + addIn[(long long)batch * sAdd + (size_t)gm * ldc + gn] + ct;
          }
        } else {
          if (gm < Mvalid && gn < Nvalid) {
            long long o = (long long)batch * sC + (size_t)gm * ldc + gn;
            outF[o] = v + addIn[(long long)batch * sAdd + (size_t)gm * ldc + gn];
          }
        }
      }
    }
  }
}

// ---------------------------------------------------------------------------
// Bidirectional LSTM recurrence.  grid = {2 directions}, block = 320 (10 waves).
// Per step: g(32,600) = pre[t] + h(32,160pad) @ W_hh(640pad,160)^T  via WMMA,
// activations staged bf16 in LDS, c kept in registers (15 elems/thread).
// Writes f32 encodings (B,T,300), bf16 transposed encodings (B,384pad,T),
// and the final hidden state.
// ---------------------------------------------------------------------------
__global__ __launch_bounds__(320)
void lstm_kernel(const float* __restrict__ pre_f, const float* __restrict__ pre_b,
                 const unsigned short* __restrict__ whh_f,
                 const unsigned short* __restrict__ whh_b,
                 float* __restrict__ enc_out, float* __restrict__ last_out,
                 unsigned short* __restrict__ encT) {
  const int dir = blockIdx.x;
  const float* pre = dir ? pre_b : pre_f;
  const unsigned short* whh = dir ? whh_b : whh_f;
  const int dirOff = dir ? 150 : 0;

  __shared__ __align__(16) unsigned short sH[32 * 160];  // h, bf16, K-padded
  __shared__ unsigned short sG[32 * 600];                // activated gates, bf16

  const int tid   = threadIdx.x;
  const int lane  = tid & 31;
  const int wave  = tid >> 5;      // 0..9, each owns 64 gate columns
  const int l16   = lane >> 4;
  const int ln    = lane & 15;
  const int nBase = wave * 64;

  for (int i = tid; i < 32 * 160; i += 320) sH[i] = 0;
  __syncthreads();

  float cR[15], hR[15];
#pragma unroll
  for (int i = 0; i < 15; ++i) { cR[i] = 0.f; hR[i] = 0.f; }
  const int eBase = tid * 15;      // 320*15 == 32*150

  for (int t = 0; t < 1024; ++t) {
    const int tt = dir ? (1023 - t) : t;

    // ---- phase 1: gates = pre[tt] + h @ W_hh^T, then activations -> sG ----
    Frag afr[2][5];
#pragma unroll
    for (int mt = 0; mt < 2; ++mt)
#pragma unroll
      for (int kk = 0; kk < 5; ++kk) {
        int base = (mt * 16 + ln) * 160 + kk * 32 + 8 * l16;
        afr[mt][kk].q[0] = *(const uint4*)&sH[base];
        afr[mt][kk].q[1] = *(const uint4*)&sH[base + 16];
      }
#pragma unroll
    for (int nt = 0; nt < 4; ++nt) {
      const int n0 = nBase + nt * 16;
      Frag bfr[5];
#pragma unroll
      for (int kk = 0; kk < 5; ++kk) {   // W_hh stays hot in L2; frags re-read each step
        const uint4* p = (const uint4*)(whh + (size_t)(n0 + ln) * 160 + kk * 32 + 8 * l16);
        bfr[kk].q[0] = p[0]; bfr[kk].q[1] = p[2];
      }
      const int n = n0 + ln;
#pragma unroll
      for (int mt = 0; mt < 2; ++mt) {
        v8f acc = {0.f, 0.f, 0.f, 0.f, 0.f, 0.f, 0.f, 0.f};
#pragma unroll
        for (int kk = 0; kk < 5; ++kk)
          acc = __builtin_amdgcn_wmma_f32_16x16x32_bf16(
              false, afr[mt][kk].v, false, bfr[kk].v, (short)0, acc, false, false);
        if (n < 600) {
          const bool isTanh = (n >= 300) && (n < 450);   // gate order: i,f,g,o
#pragma unroll
          for (int r = 0; r < 8; ++r) {
            int m = mt * 16 + r + 8 * l16;               // batch index 0..31
            float g = acc[r] + pre[((size_t)m * 1024 + tt) * 600 + n];
            float a = isTanh ? tanhf(g) : sigmoidf_(g);
            sG[m * 600 + n] = f2bf(a);
          }
        }
      }
    }
    __syncthreads();

    // ---- phase 2: c,h update (c in registers), write h everywhere ----
#pragma unroll
    for (int i = 0; i < 15; ++i) {
      int e = eBase + i;
      int m = e / 150, n = e % 150;
      float iv = bf2f(sG[m * 600 + n]);
      float fv = bf2f(sG[m * 600 + n + 150]);
      float gv = bf2f(sG[m * 600 + n + 300]);
      float ov = bf2f(sG[m * 600 + n + 450]);
      float c = fv * cR[i] + iv * gv;
      float h = ov * tanhf(c);
      cR[i] = c; hR[i] = h;
      sH[m * 160 + n] = f2bf(h);
      enc_out[((size_t)m * 1024 + tt) * 300 + dirOff + n] = h;
      encT[((size_t)m * 384 + dirOff + n) * 1024 + tt] = f2bf(h);
    }
    __syncthreads();
  }

#pragma unroll
  for (int i = 0; i < 15; ++i) {
    int e = eBase + i;
    int m = e / 150, n = e % 150;
    last_out[m * 300 + dirOff + n] = hR[i];
  }
}

// ---------------------------------------------------------------------------
extern "C" void kernel_launch(void* const* d_in, const int* in_sizes, int n_in,
                              void* d_out, int out_size, void* d_ws, size_t ws_size,
                              hipStream_t stream) {
  (void)in_sizes; (void)n_in; (void)out_size; (void)ws_size;

  const float* src   = (const float*)d_in[0];   // (32,1024,300)
  const float* tabIn = (const float*)d_in[1];   // (32,256,300)
  const float* schIn = (const float*)d_in[2];   // (32,64,300)
  const float* hot   = (const float*)d_in[3];   // (32,256,9)
  const float* Wihf  = (const float*)d_in[4];   // (600,300)
  const float* Whhf  = (const float*)d_in[5];   // (600,150)
  const float* bihf  = (const float*)d_in[6];
  const float* bhhf  = (const float*)d_in[7];
  const float* Wihb  = (const float*)d_in[8];
  const float* Whhb  = (const float*)d_in[9];
  const float* bihb  = (const float*)d_in[10];
  const float* bhhb  = (const float*)d_in[11];
  const float* colW  = (const float*)d_in[12];  // (300,9)
  const float* colb  = (const float*)d_in[13];  // (300)
  const int*   tmask = (const int*)d_in[14];    // (32,256)
  const int*   smask = (const int*)d_in[15];    // (32,64)
  float* out = (float*)d_out;

  float* outEnc  = out;                 // 32*1024*300
  float* outTab  = out + 9830400;       // 32*256*300
  float* outSch  = out + 12288000;      // 32*64*300
  float* outLast = out + 12902400;      // 32*300

  char* ws = (char*)d_ws;
  size_t off = 0;
  auto carve = [&](size_t bytes) -> char* {
    char* p = ws + off;
    off += (bytes + 255) & ~(size_t)255;
    return p;
  };
  unsigned short* snB   = (unsigned short*)carve(32768ull * 320 * 2);
  unsigned short* tnB   = (unsigned short*)carve(8192ull * 320 * 2);
  unsigned short* schnB = (unsigned short*)carve(32ull * 128 * 320 * 2);
  unsigned short* wfB   = (unsigned short*)carve(640ull * 320 * 2);
  unsigned short* wbB   = (unsigned short*)carve(640ull * 320 * 2);
  unsigned short* whfB  = (unsigned short*)carve(640ull * 160 * 2);
  unsigned short* whbB  = (unsigned short*)carve(640ull * 160 * 2);
  float*          biasB = (float*)carve(1200ull * 4);
  float*          normB = (float*)carve(32768ull * 4);
  float*          preF  = (float*)carve(32768ull * 600 * 4);
  float*          preB  = (float*)carve(32768ull * 600 * 4);
  unsigned short* simT  = (unsigned short*)carve(32ull * 256 * 1024 * 2);
  unsigned short* simS  = (unsigned short*)carve(32ull * 128 * 1024 * 2);
  unsigned short* encT  = (unsigned short*)carve(32ull * 384 * 1024 * 2);
  // total ~238 MB of d_ws

  // 1) biases + staging (bf16 casts, row norms, padding)
  bias_kernel<<<5, 256, 0, stream>>>(bihf, bhhf, bihb, bhhb, biasB);
  stage_rows_kernel<<<32768, 256, 0, stream>>>(src,   snB,   normB,  32768, 32768, 300, 320, 1);
  stage_rows_kernel<<<8192,  256, 0, stream>>>(tabIn, tnB,   nullptr, 8192,  8192, 300, 320, 1);
  stage_rows_kernel<<<4096,  256, 0, stream>>>(schIn, schnB, nullptr,   64,   128, 300, 320, 1);
  stage_rows_kernel<<<640,   256, 0, stream>>>(Wihf,  wfB,   nullptr,  600,   640, 300, 320, 0);
  stage_rows_kernel<<<640,   256, 0, stream>>>(Wihb,  wbB,   nullptr,  600,   640, 300, 320, 0);
  stage_rows_kernel<<<640,   256, 0, stream>>>(Whhf,  whfB,  nullptr,  600,   640, 150, 160, 0);
  stage_rows_kernel<<<640,   256, 0, stream>>>(Whhb,  whbB,  nullptr,  600,   640, 150, 160, 0);
  zero16_kernel<<<6144, 256, 0, stream>>>((uint4*)encT, (long long)(32ull * 384 * 1024 * 2 / 16));

  // 2) pre-projection GEMMs:  pre = (sn * norm) @ W_ih^T + bias  (un-normalize in epilogue)
  gemm_bt_kernel<<<dim3(5, 256, 1), 256, 0, stream>>>(
      snB, wfB, 320, 0, 0, EPI_BIAS_F32, preF, nullptr, 0, 600, 32768, 600,
      biasB, normB, nullptr, 0, nullptr, 0, nullptr, nullptr, nullptr);
  gemm_bt_kernel<<<dim3(5, 256, 1), 256, 0, stream>>>(
      snB, wbB, 320, 0, 0, EPI_BIAS_F32, preB, nullptr, 0, 600, 32768, 600,
      biasB + 600, normB, nullptr, 0, nullptr, 0, nullptr, nullptr, nullptr);

  // 3) cosine similarity GEMMs (masked, bf16 out)
  gemm_bt_kernel<<<dim3(8, 2, 32), 256, 0, stream>>>(
      tnB, snB, 320, 256ll * 320, 1024ll * 320, EPI_MASK_BF16, nullptr, simT,
      256ll * 1024, 1024, 256, 1024, nullptr, nullptr, tmask, 256,
      nullptr, 0, nullptr, nullptr, nullptr);
  gemm_bt_kernel<<<dim3(8, 1, 32), 256, 0, stream>>>(
      schnB, snB, 320, 128ll * 320, 1024ll * 320, EPI_MASK_BF16, nullptr, simS,
      128ll * 1024, 1024, 64, 1024, nullptr, nullptr, smask, 64,
      nullptr, 0, nullptr, nullptr, nullptr);

  // 4) sequential bidirectional LSTM recurrence (2 blocks, one per direction)
  lstm_kernel<<<2, 320, 0, stream>>>(preF, preB, whfB, whbB, outEnc, outLast, encT);

  // 5) context GEMMs: sim(C,T) @ encT(300pad384,T)^T with fused residual epilogues
  gemm_bt_kernel<<<dim3(3, 2, 32), 256, 0, stream>>>(
      simT, encT, 1024, 256ll * 1024, 384ll * 1024, EPI_ADD_TABLE, outTab, nullptr,
      256ll * 300, 300, 256, 300, nullptr, nullptr, nullptr, 0,
      tabIn, 256ll * 300, hot, colW, colb);
  gemm_bt_kernel<<<dim3(3, 1, 32), 256, 0, stream>>>(
      simS, encT, 1024, 128ll * 1024, 384ll * 1024, EPI_ADD_SCHEMA, outSch, nullptr,
      64ll * 300, 300, 64, 300, nullptr, nullptr, nullptr, 0,
      schIn, 64ll * 300, nullptr, nullptr, nullptr);
}
